// MultiHeadedAttentionWithDeCov_26190710571435
// MI455X (gfx1250) — matrix operand order repaired and checked by reading, not compiled, not run on hardware
//
#include <hip/hip_runtime.h>

// ---------------------------------------------------------------------------
// MultiHeadedAttention + DeCov for MI455X (gfx1250, wave32, WMMA bf16)
// B=4, L=1024, D=1024, H=16, DK=64
// ---------------------------------------------------------------------------

typedef __attribute__((ext_vector_type(16))) __bf16 v16bf;
typedef __attribute__((ext_vector_type(8)))  __bf16 v8bf;
typedef __attribute__((ext_vector_type(8)))  float  v8f;

#define Bq 4
#define Lq 1024
#define Dq 1024
#define Hq 16
#define DKq 64
#define Nrows 4096           // B*L
#define OUT_ELEMS 4194304    // B*L*D

__device__ __forceinline__ v8f wmma_bf16(v16bf a, v16bf b, v8f c) {
  return __builtin_amdgcn_wmma_f32_16x16x32_bf16(false, a, false, b,
                                                 (short)0, c, false, false);
}

// A-fragment: 16(M) x 32(K) bf16, src pre-offset to (row0,k0), row stride ld.
// lane = row + 16*g ; elems 0..7 -> K = 8g..8g+7 ; 8..15 -> K = 16+8g..+7
__device__ __forceinline__ v16bf load_a_frag(const __bf16* src, int ld, int lane) {
  int row = lane & 15, g = lane >> 4;
  const __bf16* p = src + row * ld + g * 8;
  v8bf lo = *(const v8bf*)(p);
  v8bf hi = *(const v8bf*)(p + 16);
  v16bf a;
#pragma unroll
  for (int i = 0; i < 8; ++i) { a[i] = lo[i]; a[i + 8] = hi[i]; }
  return a;
}

// B-fragment: 32(K) x 16(N) bf16 from "Bt" storage bt[n][k] (k contiguous),
// pre-offset to (n0,k0), row stride ld. lane = col + 16*g ; elems 0..7 -> K = 8g..8g+7 ; 8..15 -> K = 16+8g..+7
__device__ __forceinline__ v16bf load_b_frag(const __bf16* bt, int ld, int lane) {
  int col = lane & 15, g = lane >> 4;
  const __bf16* p = bt + col * ld + g * 8;
  v8bf lo = *(const v8bf*)(p);
  v8bf hi = *(const v8bf*)(p + 16);
  v16bf b;
#pragma unroll
  for (int i = 0; i < 8; ++i) { b[i] = lo[i]; b[i + 8] = hi[i]; }
  return b;
}

// ---------------------------------------------------------------------------
// Pipelined 32(M) x 64(N) x KDIM register-tile GEMM core (one wave).
// Double-buffers A/B fragments so loads for k+32 overlap WMMAs for k.
// pA pre-offset to (m0, 0); pB pre-offset to (n0, 0).
// ---------------------------------------------------------------------------
template <int KDIM>
__device__ __forceinline__ void gemm_32x64_core(const __bf16* __restrict__ pA, int lda,
                                                const __bf16* __restrict__ pB, int ldb,
                                                int lane, v8f acc[2][4]) {
  v16bf a0[2], a1[2], b0[4], b1[4];
  a0[0] = load_a_frag(pA, lda, lane);
  a0[1] = load_a_frag(pA + 16 * (size_t)lda, lda, lane);
#pragma unroll
  for (int t = 0; t < 4; ++t)
    b0[t] = load_b_frag(pB + (size_t)(16 * t) * ldb, ldb, lane);

  for (int k0 = 0; k0 < KDIM; k0 += 64) {
    // stage k0+32 into buffer 1
    {
      const __bf16* pa = pA + k0 + 32;
      const __bf16* pb = pB + k0 + 32;
      a1[0] = load_a_frag(pa, lda, lane);
      a1[1] = load_a_frag(pa + 16 * (size_t)lda, lda, lane);
#pragma unroll
      for (int t = 0; t < 4; ++t)
        b1[t] = load_b_frag(pb + (size_t)(16 * t) * ldb, ldb, lane);
      __builtin_prefetch(pa + 64, 0, 1);  // global_prefetch_b8 next tile
    }
    // compute k0 from buffer 0
#pragma unroll
    for (int t = 0; t < 4; ++t) {
      acc[0][t] = wmma_bf16(a0[0], b0[t], acc[0][t]);
      acc[1][t] = wmma_bf16(a0[1], b0[t], acc[1][t]);
    }
    // stage k0+64 into buffer 0
    if (k0 + 64 < KDIM) {
      const __bf16* pa = pA + k0 + 64;
      const __bf16* pb = pB + k0 + 64;
      a0[0] = load_a_frag(pa, lda, lane);
      a0[1] = load_a_frag(pa + 16 * (size_t)lda, lda, lane);
#pragma unroll
      for (int t = 0; t < 4; ++t)
        b0[t] = load_b_frag(pb + (size_t)(16 * t) * ldb, ldb, lane);
    }
    // compute k0+32 from buffer 1
#pragma unroll
    for (int t = 0; t < 4; ++t) {
      acc[0][t] = wmma_bf16(a1[0], b1[t], acc[0][t]);
      acc[1][t] = wmma_bf16(a1[1], b1[t], acc[1][t]);
    }
  }
}

// ---------------------------------------------------------------------------
// fp32 -> bf16 bulk convert
// ---------------------------------------------------------------------------
__global__ __launch_bounds__(256) void convert_kernel(const float* __restrict__ src,
                                                      __bf16* __restrict__ dst, int n) {
  int i = blockIdx.x * 256 + threadIdx.x;
  if (i < n) dst[i] = (__bf16)src[i];
}

// ---------------------------------------------------------------------------
// QKV projection: [4096,1024] x [1024,1024]^T (+bias). Q,K -> [B,H,L,DK] bf16;
// V -> transposed [B,H,DK,L] bf16. grid (128, 4, 3), block 128 (4 waves).
// ---------------------------------------------------------------------------
__global__ __launch_bounds__(128)
void qkv_gemm_kernel(const __bf16* __restrict__ xbf,
                     const __bf16* __restrict__ wq, const __bf16* __restrict__ wk,
                     const __bf16* __restrict__ wv,
                     const float* __restrict__ bq, const float* __restrict__ bk,
                     const float* __restrict__ bv,
                     __bf16* __restrict__ Q, __bf16* __restrict__ K,
                     __bf16* __restrict__ Vt) {
  const int which = blockIdx.z;  // 0=Q 1=K 2=V
  const __bf16* W = (which == 0) ? wq : (which == 1) ? wk : wv;
  const float* bias = (which == 0) ? bq : (which == 1) ? bk : bv;
  const int lane = threadIdx.x & 31;
  const int wave = threadIdx.x >> 5;
  const int m0 = blockIdx.x * 32;
  const int n0 = blockIdx.y * 256 + wave * 64;

  v8f acc[2][4] = {};
  gemm_32x64_core<Dq>(xbf + (size_t)m0 * Dq, Dq, W + (size_t)n0 * Dq, Dq, lane, acc);

  const int col = lane & 15, g = lane >> 4;
#pragma unroll
  for (int u = 0; u < 2; ++u) {
#pragma unroll
    for (int t = 0; t < 4; ++t) {
      int n = n0 + 16 * t + col;
      float bb = bias[n];
      int h = n >> 6, dk = n & 63;
#pragma unroll
      for (int r = 0; r < 8; ++r) {
        int m = m0 + 16 * u + r + 8 * g;
        int bi = m >> 10, l = m & 1023;
        float val = acc[u][t][r] + bb;
        if (which == 2) {
          Vt[(((size_t)(bi * Hq + h) * DKq + dk) << 10) + l] = (__bf16)val;
        } else {
          __bf16* dst = (which == 0) ? Q : K;
          dst[(((size_t)(bi * Hq + h) << 10) + l) * DKq + dk] = (__bf16)val;
        }
      }
    }
  }
}

// ---------------------------------------------------------------------------
// Flash attention: one wave handles 16 query rows of one (b,h); online softmax.
// grid (B*H=64, L/16=64), block 32. Output merged bf16 [B*L, D].
// ---------------------------------------------------------------------------
__global__ __launch_bounds__(32)
void attn_kernel(const __bf16* __restrict__ Q, const __bf16* __restrict__ K,
                 const __bf16* __restrict__ Vt, __bf16* __restrict__ merged) {
  __shared__ __attribute__((aligned(16))) __bf16 pls[16 * 32];  // P tile, C->A relayout
  const int bh = blockIdx.x;
  const int m0 = blockIdx.y * 16;
  const int lane = threadIdx.x & 31;
  const int b = bh >> 4, h = bh & 15;
  const __bf16* Qb = Q + (size_t)bh * Lq * DKq;
  const __bf16* Kb = K + (size_t)bh * Lq * DKq;
  const __bf16* Vb = Vt + (size_t)bh * DKq * Lq;

  const v16bf qa0 = load_a_frag(Qb + m0 * DKq + 0, DKq, lane);
  const v16bf qa1 = load_a_frag(Qb + m0 * DKq + 32, DKq, lane);

  float mrow[8], lrow[8];
  v8f acc[4] = {};
#pragma unroll
  for (int r = 0; r < 8; ++r) { mrow[r] = -1e30f; lrow[r] = 0.f; }
  const int col = lane & 15, g = lane >> 4;

  for (int n0 = 0; n0 < Lq; n0 += 32) {
    v8f s0 = {}, s1 = {};
    s0 = wmma_bf16(qa0, load_b_frag(Kb + (size_t)n0 * DKq + 0, DKq, lane), s0);
    s0 = wmma_bf16(qa1, load_b_frag(Kb + (size_t)n0 * DKq + 32, DKq, lane), s0);
    s1 = wmma_bf16(qa0, load_b_frag(Kb + (size_t)(n0 + 16) * DKq + 0, DKq, lane), s1);
    s1 = wmma_bf16(qa1, load_b_frag(Kb + (size_t)(n0 + 16) * DKq + 32, DKq, lane), s1);

#pragma unroll
    for (int r = 0; r < 8; ++r) {
      float a0 = s0[r] * 0.125f;   // 1/sqrt(64)
      float a1 = s1[r] * 0.125f;
      float tmax = fmaxf(a0, a1);
#pragma unroll
      for (int off = 1; off < 16; off <<= 1)
        tmax = fmaxf(tmax, __shfl_xor(tmax, off, 16));
      float nm = fmaxf(mrow[r], tmax);
      float p0 = __expf(a0 - nm), p1 = __expf(a1 - nm);
      float rs = p0 + p1;
#pragma unroll
      for (int off = 1; off < 16; off <<= 1) rs += __shfl_xor(rs, off, 16);
      float alpha = __expf(mrow[r] - nm);
      lrow[r] = lrow[r] * alpha + rs;
      mrow[r] = nm;
#pragma unroll
      for (int t = 0; t < 4; ++t) acc[t][r] = acc[t][r] * alpha;
      int row = r + 8 * g;
      pls[row * 32 + col] = (__bf16)p0;
      pls[row * 32 + 16 + col] = (__bf16)p1;
    }
    __syncthreads();  // single-wave block: ds-count wait
    v16bf pa = load_a_frag(pls, 32, lane);
#pragma unroll
    for (int t = 0; t < 4; ++t) {
      v16bf vb = load_b_frag(Vb + (size_t)(16 * t) * Lq + n0, Lq, lane);
      acc[t] = wmma_bf16(pa, vb, acc[t]);
    }
    __syncthreads();
  }

#pragma unroll
  for (int r = 0; r < 8; ++r) {
    float inv = 1.f / lrow[r];
    int l = m0 + r + 8 * g;
#pragma unroll
    for (int t = 0; t < 4; ++t) {
      int d = h * DKq + t * 16 + col;
      merged[((size_t)(b * Lq + l)) * Dq + d] = (__bf16)(acc[t][r] * inv);
    }
  }
}

// ---------------------------------------------------------------------------
// Column mean of merged [4096, 1024] -> mean[1024]
// ---------------------------------------------------------------------------
__global__ __launch_bounds__(256)
void colmean_kernel(const __bf16* __restrict__ merged, float* __restrict__ mean) {
  __shared__ float red[256];
  const int d = blockIdx.x;
  float s = 0.f;
  for (int m = threadIdx.x; m < Nrows; m += 256)
    s += (float)merged[(size_t)m * Dq + d];
  red[threadIdx.x] = s;
  __syncthreads();
  for (int off = 128; off > 0; off >>= 1) {
    if ((int)threadIdx.x < off) red[threadIdx.x] += red[threadIdx.x + off];
    __syncthreads();
  }
  if (threadIdx.x == 0) mean[d] = red[0] * (1.f / (float)Nrows);
}

// ---------------------------------------------------------------------------
// Ut[d][m] = merged[m][d] - mean[d] (bf16, transposed for DeCov GEMM);
// also zero the decov accumulator.
// ---------------------------------------------------------------------------
__global__ __launch_bounds__(256)
void build_ut_kernel(const __bf16* __restrict__ merged, const float* __restrict__ mean,
                     __bf16* __restrict__ ut, float* __restrict__ decov_out) {
  size_t idx = (size_t)blockIdx.x * 256 + threadIdx.x;
  if (idx == 0) *decov_out = 0.f;
  int d = (int)(idx >> 12);      // /4096
  int m = (int)(idx & 4095);
  ut[idx] = (__bf16)((float)merged[(size_t)m * Dq + d] - mean[d]);
}

// ---------------------------------------------------------------------------
// Output projection: out = merged @ Wo^T + bo (fp32 out). grid (128,4), block 128.
// ---------------------------------------------------------------------------
__global__ __launch_bounds__(128)
void out_gemm_kernel(const __bf16* __restrict__ merged, const __bf16* __restrict__ wo,
                     const float* __restrict__ bo, float* __restrict__ out) {
  const int lane = threadIdx.x & 31;
  const int wave = threadIdx.x >> 5;
  const int m0 = blockIdx.x * 32;
  const int n0 = blockIdx.y * 256 + wave * 64;

  v8f acc[2][4] = {};
  gemm_32x64_core<Dq>(merged + (size_t)m0 * Dq, Dq, wo + (size_t)n0 * Dq, Dq, lane, acc);

  const int col = lane & 15, g = lane >> 4;
#pragma unroll
  for (int u = 0; u < 2; ++u)
#pragma unroll
    for (int t = 0; t < 4; ++t) {
      int n = n0 + 16 * t + col;
      float bb = bo[n];
#pragma unroll
      for (int r = 0; r < 8; ++r)
        out[(size_t)(m0 + 16 * u + r + 8 * g) * Dq + n] = acc[u][t][r] + bb;
    }
}

// ---------------------------------------------------------------------------
// DeCov: C = Ut x Ut^T / N ; accumulate 0.5 * sum_{i!=j} C_ij^2.
// grid (32, 16), block 32; each wave: 32x64 tile of C, K=4096.
// ---------------------------------------------------------------------------
__global__ __launch_bounds__(32)
void decov_kernel(const __bf16* __restrict__ ut, float* __restrict__ decov_out) {
  const int lane = threadIdx.x & 31;
  const int i0 = blockIdx.x * 32;
  const int j0 = blockIdx.y * 64;

  v8f acc[2][4] = {};
  gemm_32x64_core<Nrows>(ut + (size_t)i0 * Nrows, Nrows,
                         ut + (size_t)j0 * Nrows, Nrows, lane, acc);

  const float invN = 1.f / (float)Nrows;
  const int col = lane & 15, g = lane >> 4;
  float local = 0.f;
#pragma unroll
  for (int u = 0; u < 2; ++u)
#pragma unroll
    for (int t = 0; t < 4; ++t)
#pragma unroll
      for (int r = 0; r < 8; ++r) {
        int i = i0 + 16 * u + r + 8 * g, j = j0 + 16 * t + col;
        float c = acc[u][t][r] * invN;
        if (i != j) local += c * c;
      }
#pragma unroll
  for (int off = 1; off < 32; off <<= 1) local += __shfl_xor(local, off, 32);
  if (lane == 0) atomicAdd(decov_out, 0.5f * local);
}

// ---------------------------------------------------------------------------
// Host-side orchestration
// ---------------------------------------------------------------------------
extern "C" void kernel_launch(void* const* d_in, const int* in_sizes, int n_in,
                              void* d_out, int out_size, void* d_ws, size_t ws_size,
                              hipStream_t stream) {
  const float* x  = (const float*)d_in[0];
  const float* Wq = (const float*)d_in[1];
  const float* bq = (const float*)d_in[2];
  const float* Wk = (const float*)d_in[3];
  const float* bk = (const float*)d_in[4];
  const float* Wv = (const float*)d_in[5];
  const float* bv = (const float*)d_in[6];
  const float* Wo = (const float*)d_in[7];
  const float* bo = (const float*)d_in[8];
  float* out = (float*)d_out;

  char* ws = (char*)d_ws;
  const size_t MB = 1u << 20;
  __bf16* xbf = (__bf16*)(ws + 0 * MB);     // 8 MB
  __bf16* wqb = (__bf16*)(ws + 8 * MB);     // 2 MB
  __bf16* wkb = (__bf16*)(ws + 10 * MB);    // 2 MB
  __bf16* wvb = (__bf16*)(ws + 12 * MB);    // 2 MB
  __bf16* wob = (__bf16*)(ws + 14 * MB);    // 2 MB
  __bf16* Qb  = (__bf16*)(ws + 16 * MB);    // 8 MB  [B,H,L,DK]
  __bf16* Kb  = (__bf16*)(ws + 24 * MB);    // 8 MB  [B,H,L,DK]
  __bf16* Vtb = (__bf16*)(ws + 32 * MB);    // 8 MB  [B,H,DK,L]
  __bf16* mg  = (__bf16*)(ws + 40 * MB);    // 8 MB  merged [B*L, D]
  __bf16* utb = (__bf16*)(ws + 48 * MB);    // 8 MB  Ut [D, B*L]
  float*  mean = (float*)(ws + 56 * MB);    // 4 KB
  float*  decov = out + OUT_ELEMS;          // scalar tail of d_out

  // 1) bf16 conversion
  convert_kernel<<<OUT_ELEMS / 256, 256, 0, stream>>>(x, xbf, OUT_ELEMS);
  convert_kernel<<<(Dq * Dq) / 256, 256, 0, stream>>>(Wq, wqb, Dq * Dq);
  convert_kernel<<<(Dq * Dq) / 256, 256, 0, stream>>>(Wk, wkb, Dq * Dq);
  convert_kernel<<<(Dq * Dq) / 256, 256, 0, stream>>>(Wv, wvb, Dq * Dq);
  convert_kernel<<<(Dq * Dq) / 256, 256, 0, stream>>>(Wo, wob, Dq * Dq);

  // 2) QKV projections
  qkv_gemm_kernel<<<dim3(Nrows / 32, 4, 3), 128, 0, stream>>>(
      xbf, wqb, wkb, wvb, bq, bk, bv, Qb, Kb, Vtb);

  // 3) flash attention -> merged
  attn_kernel<<<dim3(Bq * Hq, Lq / 16), 32, 0, stream>>>(Qb, Kb, Vtb, mg);

  // 4) column mean for DeCov centering
  colmean_kernel<<<Dq, 256, 0, stream>>>(mg, mean);

  // 5) build centered transposed U (and zero decov accumulator)
  build_ut_kernel<<<OUT_ELEMS / 256, 256, 0, stream>>>(mg, mean, utb, decov);

  // 6) output projection -> d_out[0 .. 4M)
  out_gemm_kernel<<<dim3(Nrows / 32, 4), 128, 0, stream>>>(mg, wob, bo, out);

  // 7) DeCov loss -> d_out[4M]
  decov_kernel<<<dim3(Dq / 32, Dq / 64), 32, 0, stream>>>(utb, decov);
}
